// MultiHeadAttention_60404420051661
// MI455X (gfx1250) — compile-verified, matrix operand order checked by reading, not run on hardware
//
#include <hip/hip_runtime.h>
#include <hip/hip_bf16.h>

typedef __attribute__((ext_vector_type(16))) __bf16 v16bf;
typedef __attribute__((ext_vector_type(8)))  __bf16 v8bf;
typedef __attribute__((ext_vector_type(4)))  __bf16 v4bf;
typedef __attribute__((ext_vector_type(8)))  float  v8f;
typedef __attribute__((ext_vector_type(4)))  unsigned int v4u;
typedef __attribute__((ext_vector_type(8)))  int v8i;
typedef __attribute__((ext_vector_type(4)))  int v4i;

static constexpr int Bsz = 4;
static constexpr int Sseq = 2048;
static constexpr int Dmodel = 1024;
static constexpr int Hh = 16;

#define WMMA_BF16(a, b, c) \
  __builtin_amdgcn_wmma_f32_16x16x32_bf16(false, (a), false, (b), (short)0, (c), false, false)

// ---------------- weight pre-convert: f32 -> bf16, 4 elems/thread ----------
__global__ __launch_bounds__(256)
void cvt_bf16_kernel(const float* __restrict__ x, __bf16* __restrict__ y) {
  int i = blockIdx.x * 256 + threadIdx.x;
  float4 v = ((const float4*)x)[i];
  v4bf o;
  o[0] = (__bf16)v.x; o[1] = (__bf16)v.y; o[2] = (__bf16)v.z; o[3] = (__bf16)v.w;
  *(v4bf*)(y + 4 * (size_t)i) = o;
}

// ---------------- projection: out[s,hd] = X[s,:] . W[hd,:] + bias ----------
// 64 rows/block, 4 row-tiles per wave share one W B-operand load.
// Double-buffered LDS staging: X prefetch overlaps the 4 WMMAs.
__global__ __launch_bounds__(256)
void proj_kernel(const float* __restrict__ X, const __bf16* __restrict__ Wb,
                 const float* __restrict__ bias, __bf16* __restrict__ out,
                 int transpose_out)
{
  __shared__ __bf16 lds_a[2][64][40];   // ping-pong 64x32 X tiles (bf16, padded)

  const int tid  = threadIdx.x;
  const int lane = tid & 31;
  const int wave = tid >> 5;
  const int row0 = blockIdx.x * 64;                 // rows in [0, B*S)
  const int col0 = (blockIdx.y * 8 + wave) * 16;    // col in [0, H*64)
  const int m  = lane & 15;
  const int kh = lane >> 4;

  const int sr = tid >> 2;            // staging: row 0..63
  const int sc = (tid & 3) * 8;       // staging: col 0,8,16,24
  const float* Xrow = X + (size_t)(row0 + sr) * Dmodel + sc;

  auto stage = [&](int buf, int k0) {
    float4 f0 = *(const float4*)(Xrow + k0);
    float4 f1 = *(const float4*)(Xrow + k0 + 4);
    v8bf pk;
    pk[0] = (__bf16)f0.x; pk[1] = (__bf16)f0.y; pk[2] = (__bf16)f0.z; pk[3] = (__bf16)f0.w;
    pk[4] = (__bf16)f1.x; pk[5] = (__bf16)f1.y; pk[6] = (__bf16)f1.z; pk[7] = (__bf16)f1.w;
    *(v8bf*)(&lds_a[buf][sr][sc]) = pk;
  };

  stage(0, 0);
  v8f acc[4] = {};
  for (int k0 = 0, it = 0; k0 < Dmodel; k0 += 32, ++it) {
    const int cur = it & 1;
    __syncthreads();                       // buf[cur] staged; buf[cur^1] free
    if (k0 + 32 < Dmodel) stage(cur ^ 1, k0 + 32);

    v16bf b = *(const v16bf*)(Wb + (size_t)(col0 + m) * Dmodel + k0 + kh * 16);
#pragma unroll
    for (int rt = 0; rt < 4; ++rt) {
      v16bf a = *(const v16bf*)(&lds_a[cur][rt * 16 + m][kh * 16]);
      acc[rt] = WMMA_BF16(a, b, acc[rt]);
    }
  }

  const float bv = bias[col0 + m];
  const int bidx = row0 / Sseq;
  const int s0   = row0 % Sseq;
  const int hd   = col0 + m;
  const int h    = hd >> 6, d = hd & 63;
#pragma unroll
  for (int rt = 0; rt < 4; ++rt) {
#pragma unroll
    for (int r = 0; r < 8; ++r) {
      int s = s0 + rt * 16 + r + 8 * kh;
      float val = acc[rt][r] + bv;
      size_t idx = transpose_out
          ? ((size_t)(bidx * Hh + h) * 64 + d) * Sseq + s
          : ((size_t)(bidx * Hh + h) * Sseq + s) * 64 + d;
      out[idx] = (__bf16)val;
    }
  }
}

// ---------------- TDM staging (gfx1250 Tensor Data Mover) ------------------
#if __has_builtin(__builtin_amdgcn_tensor_load_to_lds)
#define USE_TDM 1
// 2D tile load: global (row-major, stride0 elems between rows, bf16) -> LDS
// with per-row padding (pad codes per CDNA5 D# spec).
static __device__ __forceinline__ void tdm_load_2d(
    const __bf16* gsrc, unsigned lds_off,
    unsigned tdim0, unsigned tdim1, unsigned long long stride0,
    unsigned tile0, unsigned tile1,
    unsigned pad_int_code, unsigned pad_amt_code)
{
  unsigned long long ga = (unsigned long long)(const void*)gsrc;
  v4u g0;
  g0[0] = 1u;                                            // count=1, user desc
  g0[1] = lds_off;                                       // lds_addr
  g0[2] = (unsigned)ga;                                  // global_addr[31:0]
  g0[3] = (unsigned)((ga >> 32) & 0x01ffffffu) | (2u << 30); // addr[56:32]|type=2
  v8i g1;
  g1[0] = (int)((1u << 16) |                             // data_size = 2 bytes
                (1u << 20) |                             // pad_enable
                (pad_int_code << 22) | (pad_amt_code << 25));
  g1[1] = (int)((tdim0 & 0xffffu) << 16);                // tensor_dim0[15:0]
  g1[2] = (int)((tdim0 >> 16) | ((tdim1 & 0xffffu) << 16));
  g1[3] = (int)((tdim1 >> 16) | (tile0 << 16));          // tile_dim0
  g1[4] = (int)(tile1 & 0xffffu);                        // tile_dim1 (tile_dim2=0)
  g1[5] = (int)(unsigned)stride0;                        // tensor_dim0_stride lo
  g1[6] = (int)(unsigned)(stride0 >> 32);                // stride hi
  g1[7] = 0;
  v4i z4 = {};
#if __clang_major__ >= 23
  v8i z8 = {};
  __builtin_amdgcn_tensor_load_to_lds(g0, g1, z4, z4, z8, 0);
#else
  __builtin_amdgcn_tensor_load_to_lds(g0, g1, z4, z4, 0);
#endif
}
#endif

// ---------------- streaming fused attention --------------------------------
// 4 waves/block, each wave owns 32 query rows (two 16-row WMMA tiles).
// Raw-exp softmax (no max subtraction needed), multiplicative mask post-exp.
// K/Vt tiles double-buffered: TDM (or copy) prefetch overlaps the 16 WMMAs.
__global__ __launch_bounds__(128)
void attn_kernel(const __bf16* __restrict__ qg, const __bf16* __restrict__ kg,
                 const __bf16* __restrict__ vtg, const int* __restrict__ mask,
                 float* __restrict__ out)
{
  __shared__ __bf16 lds_k[2][32][72];       // 32 keys x 64 d (padded)
  __shared__ __bf16 lds_vt[2][64][40];      // 64 d x 32 keys (padded)
  __shared__ __bf16 lds_p[4][2][16][40];    // per-wave/tile P 16q x 32k (padded)
  __shared__ float  lds_rs[4][2][16];

  const int tid  = threadIdx.x;
  const int lane = tid & 31;
  const int wave = tid >> 5;
  const int bh = blockIdx.y;
  const int b = bh / Hh, h = bh % Hh;
  const int qrow0_blk = blockIdx.x * 128;
  const int qrow0 = qrow0_blk + wave * 32;
  const int n  = lane & 15;
  const int kh = lane >> 4;

  const __bf16* qb  = qg  + (size_t)bh * Sseq * 64;
  const __bf16* kb  = kg  + (size_t)bh * Sseq * 64;
  const __bf16* vtb = vtg + (size_t)bh * 64 * Sseq;

  v16bf qlo[2], qhi[2];
#pragma unroll
  for (int t = 0; t < 2; ++t) {
    qlo[t] = *(const v16bf*)(qb + (size_t)(qrow0 + t * 16 + n) * 64 + kh * 16);
    qhi[t] = *(const v16bf*)(qb + (size_t)(qrow0 + t * 16 + n) * 64 + 32 + kh * 16);
  }

  v8f acc[2][4] = {};
  float rsum[2][8] = {};

#ifdef USE_TDM
  auto stage_tiles = [&](int buf, int j0) {
    if (wave == 0) {
      // K tile: rows of 64 bf16 (32 DW) + 16B pad (4 DW): codes 4, 3
      tdm_load_2d(kb + (size_t)j0 * 64, (unsigned)(size_t)&lds_k[buf][0][0],
                  64, Sseq, 64, 64, 32, 4, 3);
      // Vt tile: rows of 32 bf16 (16 DW) + 16B pad (4 DW): codes 3, 3
      tdm_load_2d(vtb + j0, (unsigned)(size_t)&lds_vt[buf][0][0],
                  Sseq, 64, Sseq, 32, 64, 3, 3);
    }
  };
#else
  auto stage_tiles = [&](int buf, int j0) {
    int kr = tid >> 2, kc = (tid & 3) * 16;
    *(uint4*)(&lds_k[buf][kr][kc]) =
        *(const uint4*)(kb + (size_t)(j0 + kr) * 64 + kc);
    int dr = tid >> 1, dc = (tid & 1) * 16;
    *(uint4*)(&lds_vt[buf][dr][dc]) =
        *(const uint4*)(vtb + (size_t)dr * Sseq + j0 + dc);
  };
#endif

  const int jmax = qrow0_blk + 128;                 // causal bound for block
  stage_tiles(0, 0);
  for (int j0 = 0, it = 0; j0 < jmax; j0 += 32, ++it) {
    const int cur = it & 1;
#ifdef USE_TDM
    if (wave == 0) __builtin_amdgcn_s_wait_tensorcnt(0);   // buf[cur] landed
#endif
    __syncthreads();             // buf[cur] visible; buf[cur^1] no longer read
    if (j0 + 32 < jmax) stage_tiles(cur ^ 1, j0 + 32);     // async prefetch

    if (j0 <= qrow0 + 31) {      // wave-uniform causal skip (EXEC stays all-1)
#pragma unroll
      for (int jj = 0; jj < 2; ++jj) {
        int j = j0 + jj * 16 + n;
        float mval = (float)mask[b * Sseq + j];
        v16bf bk_lo = *(const v16bf*)(&lds_k[cur][jj * 16 + n][kh * 16]);
        v16bf bk_hi = *(const v16bf*)(&lds_k[cur][jj * 16 + n][32 + kh * 16]);
#pragma unroll
        for (int t = 0; t < 2; ++t) {
          v8f sc = {};
          sc = WMMA_BF16(qlo[t], bk_lo, sc);
          sc = WMMA_BF16(qhi[t], bk_hi, sc);
#pragma unroll
          for (int r = 0; r < 8; ++r) {
            int i = qrow0 + t * 16 + r + 8 * kh;
            float p = __expf(sc[r] * 0.125f);          // 1/sqrt(64)
            p = (j <= i) ? p * mval : 0.f;
            rsum[t][r] += p;
            lds_p[wave][t][r + 8 * kh][jj * 16 + n] = (__bf16)p;
          }
        }
      }
      asm volatile("s_wait_dscnt 0" ::: "memory");     // per-wave LDS RAW

      v16bf av0 = *(const v16bf*)(&lds_vt[cur][ 0 + n][kh * 16]);
      v16bf av1 = *(const v16bf*)(&lds_vt[cur][16 + n][kh * 16]);
      v16bf av2 = *(const v16bf*)(&lds_vt[cur][32 + n][kh * 16]);
      v16bf av3 = *(const v16bf*)(&lds_vt[cur][48 + n][kh * 16]);
#pragma unroll
      for (int t = 0; t < 2; ++t) {
        v16bf bp = *(const v16bf*)(&lds_p[wave][t][n][kh * 16]);
        acc[t][0] = WMMA_BF16(av0, bp, acc[t][0]);
        acc[t][1] = WMMA_BF16(av1, bp, acc[t][1]);
        acc[t][2] = WMMA_BF16(av2, bp, acc[t][2]);
        acc[t][3] = WMMA_BF16(av3, bp, acc[t][3]);
      }
    }
  }

  // rowsum reduction across 16-lane groups, broadcast via LDS, write output
#pragma unroll
  for (int t = 0; t < 2; ++t) {
#pragma unroll
    for (int off = 1; off < 16; off <<= 1) {
#pragma unroll
      for (int r = 0; r < 8; ++r) rsum[t][r] += __shfl_xor(rsum[t][r], off, 32);
    }
#pragma unroll
    for (int r = 0; r < 8; ++r) lds_rs[wave][t][r + 8 * kh] = rsum[t][r];
  }
  asm volatile("s_wait_dscnt 0" ::: "memory");
#pragma unroll
  for (int t = 0; t < 2; ++t) {
    float denom = lds_rs[wave][t][n] + 1e-8f;
    int s = qrow0 + t * 16 + n;
    float* op = out + (size_t)(b * Sseq + s) * (Hh * 64) + h * 64;
#pragma unroll
    for (int r = 0; r < 8; ++r) {
      int M = r + 8 * kh;
      __builtin_nontemporal_store(acc[t][0][r] / denom, op +  0 + M);
      __builtin_nontemporal_store(acc[t][1][r] / denom, op + 16 + M);
      __builtin_nontemporal_store(acc[t][2][r] / denom, op + 32 + M);
      __builtin_nontemporal_store(acc[t][3][r] / denom, op + 48 + M);
    }
  }
}

extern "C" void kernel_launch(void* const* d_in, const int* in_sizes, int n_in,
                              void* d_out, int out_size, void* d_ws, size_t ws_size,
                              hipStream_t stream) {
  (void)in_sizes; (void)n_in; (void)out_size; (void)ws_size;
  const float* Q   = (const float*)d_in[0];
  const float* K   = (const float*)d_in[1];
  const float* V   = (const float*)d_in[2];
  const int*   msk = (const int*)d_in[3];
  const float* W_Q = (const float*)d_in[4];
  const float* W_K = (const float*)d_in[5];
  const float* W_V = (const float*)d_in[6];
  const float* b_Q = (const float*)d_in[7];
  const float* b_K = (const float*)d_in[8];
  const float* b_V = (const float*)d_in[9];
  float* out = (float*)d_out;

  const size_t per  = (size_t)Bsz * Hh * Sseq * 64;      // bf16 elems per tensor
  const size_t wper = (size_t)Dmodel * Hh * 64;          // 1M bf16 per weight
  __bf16* wq  = (__bf16*)d_ws;
  __bf16* wk  = wq + per;
  __bf16* wvt = wk + per;
  __bf16* wWq = wvt + per;
  __bf16* wWk = wWq + wper;
  __bf16* wWv = wWk + wper;

  // weights f32 -> bf16 (1M elems each, 4/thread, 256/block)
  cvt_bf16_kernel<<<wper / 1024, 256, 0, stream>>>(W_Q, wWq);
  cvt_bf16_kernel<<<wper / 1024, 256, 0, stream>>>(W_K, wWk);
  cvt_bf16_kernel<<<wper / 1024, 256, 0, stream>>>(W_V, wWv);

  dim3 pg(Bsz * Sseq / 64, (Hh * 64) / (16 * 8));        // 128 x 8
  proj_kernel<<<pg, 256, 0, stream>>>(Q, wWq, b_Q, wq, 0);
  proj_kernel<<<pg, 256, 0, stream>>>(K, wWk, b_K, wk, 0);
  proj_kernel<<<pg, 256, 0, stream>>>(V, wWv, b_V, wvt, 1);

  dim3 ag(Sseq / 128, Bsz * Hh);                         // 16 x 64
  attn_kernel<<<ag, 128, 0, stream>>>(wq, wk, wvt, msk, out);
}
